// NeighborAttentionEncoder_52553219834578
// MI455X (gfx1250) — matrix operand order repaired and checked by reading, not compiled, not run on hardware
//
#include <hip/hip_runtime.h>
#include <hip/hip_bf16.h>
#include <stdint.h>

#define D_MODEL   2560
#define NUM_HEADS 8
#define HEAD_DIM  320
#define NUM_LAYERS 2
#define D_FF      1024
#define NB        1024
#define NK        32
#define M_TOTAL   (NB * NK)      // 32768 rows
#define LN_EPS    1e-5f

typedef __attribute__((ext_vector_type(16))) __bf16 v16bf;
typedef __attribute__((ext_vector_type(8)))  float  v8f;

union FragBF { v16bf v; uint4 q[2]; };

#if defined(__has_builtin)
#  if __has_builtin(__builtin_amdgcn_global_load_async_to_lds_b128)
#    define HAVE_ASYNC_LDS 1
#  endif
#  if __has_builtin(__builtin_amdgcn_s_wait_asynccnt)
#    define HAVE_WAIT_ASYNC 1
#  endif
#endif

#if defined(HAVE_ASYNC_LDS)
typedef __attribute__((ext_vector_type(4))) int v4i;
typedef __attribute__((address_space(1))) v4i g_v4i;   // global int4
typedef __attribute__((address_space(3))) v4i l_v4i;   // LDS int4
#endif

__device__ __forceinline__ void wait_async_zero() {
#if defined(HAVE_ASYNC_LDS)
#  if defined(HAVE_WAIT_ASYNC)
    __builtin_amdgcn_s_wait_asynccnt(0);
#  else
    asm volatile("s_wait_asynccnt 0x0" ::: "memory");
#  endif
#endif
}

__device__ __forceinline__ unsigned short f2bf(float f) {
    union { float f; unsigned u; } c; c.f = f;
    unsigned u = c.u;
    u += 0x7FFFu + ((u >> 16) & 1u);   // round-to-nearest-even
    return (unsigned short)(u >> 16);
}
__device__ __forceinline__ float bf2f(unsigned short h) {
    union { unsigned u; float f; } c; c.u = ((unsigned)h) << 16;
    return c.f;
}

// ---------------------------------------------------------------------------
// X init: f32 copy (residual stream) + bf16 mirror (WMMA A operand)
// ---------------------------------------------------------------------------
__global__ __launch_bounds__(256) void cast_x_kernel(
    const float* __restrict__ x, float* __restrict__ xf,
    unsigned short* __restrict__ xbf, long long n)
{
    long long i = (long long)blockIdx.x * 256 + threadIdx.x;
    if (i >= n) return;
    float v = x[i];
    xf[i]  = v;
    xbf[i] = f2bf(v);
}

// ---------------------------------------------------------------------------
// Weight cast+transpose: W[K][N] f32 -> Wt[N][K] bf16 (contiguous B-fragments)
// ---------------------------------------------------------------------------
__global__ __launch_bounds__(256) void wcast_kernel(
    const float* __restrict__ W, unsigned short* __restrict__ Wt,
    int Kd, int N)
{
    long long i = (long long)blockIdx.x * 256 + threadIdx.x;
    long long total = (long long)Kd * N;
    if (i >= total) return;
    int n = (int)(i / Kd);
    int k = (int)(i % Kd);
    Wt[i] = f2bf(W[(long long)k * N + n]);
}

// ---------------------------------------------------------------------------
// Distance-bias MLP: per (b,k): relu(d*W1+b1) @ W2 + b2  -> db[b][h][k]
// ---------------------------------------------------------------------------
__global__ __launch_bounds__(256) void dist_bias_kernel(
    const float* __restrict__ dist,
    const float* __restrict__ W1, const float* __restrict__ b1,
    const float* __restrict__ W2, const float* __restrict__ b2,
    float* __restrict__ db)
{
    int i = blockIdx.x * 256 + threadIdx.x;
    if (i >= NB * NK) return;
    int b = i / NK, k = i % NK;
    float d = dist[i];
    float acc[NUM_HEADS];
#pragma unroll
    for (int h = 0; h < NUM_HEADS; ++h) acc[h] = b2[h];
    for (int j = 0; j < 64; ++j) {
        float hj = fmaxf(d * W1[j] + b1[j], 0.0f);
#pragma unroll
        for (int h = 0; h < NUM_HEADS; ++h) acc[h] += hj * W2[j * NUM_HEADS + h];
    }
#pragma unroll
    for (int h = 0; h < NUM_HEADS; ++h)
        db[((long long)b * NUM_HEADS + h) * NK + k] = acc[h];
}

// ---------------------------------------------------------------------------
// bf16 WMMA GEMM:  out[M][N] = act( A[M][Kd] @ Wt[N][Kd]^T + bias[N] )
// Block: 256 threads = 8 waves; block tile 64(M) x 256(N), K-stage 64.
// Wave tile 32(M) x 64(N): 2 A-frags x 4 B-frags = 8 WMMAs per 32-K substep.
// A tile double-buffered in LDS, filled by GLOBAL_LOAD_ASYNC_TO_LDS_B128
// (ASYNCcnt) when available; weight stream prefetched into cache.
// ---------------------------------------------------------------------------
#define GBM 64
#define GBN 256
#define GBK 64

template <int ACT>   // 0 = none, 1 = relu
__global__ __launch_bounds__(256) void gemm_bf16_wmma(
    const unsigned short* __restrict__ A,
    const unsigned short* __restrict__ Wt,
    const float* __restrict__ bias,
    unsigned short* __restrict__ out,
    int M, int N, int Kd)
{
    __shared__ unsigned short As[2][GBM * GBK];   // 2 x 8 KB

    const int tid   = threadIdx.x;
    const int lane  = tid & 31;
    const int wave  = tid >> 5;
    const int nwave = wave & 3;   // 4 N-waves x 64 cols
    const int mwave = wave >> 2;  // 2 M-waves x 32 rows
    const int hi    = lane >> 4;  // lane half (wave32)
    const int ml    = lane & 15;

    const int blockM = blockIdx.y * GBM;
    const int blockN = blockIdx.x * GBN;

    int nc[4];
#pragma unroll
    for (int t = 0; t < 4; ++t) nc[t] = blockN + nwave * 64 + t * 16 + ml;

    v8f zero;
#pragma unroll
    for (int j = 0; j < 8; ++j) zero[j] = 0.0f;
    v8f acc[2][4];
#pragma unroll
    for (int mt = 0; mt < 2; ++mt)
#pragma unroll
        for (int t = 0; t < 4; ++t) acc[mt][t] = zero;

    // tile fill: 64 rows x 64 K bf16 = 8 KB = 512 uint4; 2 per thread
    auto load_tile = [&](int kk, int buf) {
#pragma unroll
        for (int r = 0; r < 2; ++r) {
            int lin = tid + r * 256;
            int row = lin >> 3;
            int ch  = (lin & 7) * 8;
            const unsigned short* gp = A + (long long)(blockM + row) * Kd + kk + ch;
            unsigned short*       lp = &As[buf][row * GBK + ch];
#if defined(HAVE_ASYNC_LDS)
            __builtin_amdgcn_global_load_async_to_lds_b128(
                (g_v4i*)gp, (l_v4i*)lp, 0, 0);
#else
            *(uint4*)lp = *(const uint4*)gp;
#endif
        }
    };

    const int nIter = Kd / GBK;

    load_tile(0, 0);
    wait_async_zero();
    __syncthreads();

    for (int it = 0; it < nIter; ++it) {
        const int cur = it & 1;
        const int kk  = it * GBK;
        if (it + 1 < nIter) load_tile(kk + GBK, 1 - cur);

#pragma unroll
        for (int ks = 0; ks < GBK; ks += 32) {
            // A fragments: lane half picks 16 contiguous K (0-15 / 16-31), matching B
            FragBF a[2];
#pragma unroll
            for (int mt = 0; mt < 2; ++mt) {
                const unsigned short* arow =
                    &As[cur][(mwave * 32 + mt * 16 + ml) * GBK + ks];
                a[mt].q[0] = *(const uint4*)(arow + hi * 16);
                a[mt].q[1] = *(const uint4*)(arow + hi * 16 + 8);
            }
#pragma unroll
            for (int t = 0; t < 4; ++t) {
                // B fragment: lane holds 16 consecutive K of column nc[t] (32B)
                const unsigned short* wrow =
                    Wt + (long long)nc[t] * Kd + kk + ks + hi * 16;
                FragBF bfr;
                bfr.q[0] = ((const uint4*)wrow)[0];
                bfr.q[1] = ((const uint4*)wrow)[1];
                if (it + 1 < nIter)
                    __builtin_prefetch(wrow + GBK, 0, 1);   // next K-slab of W
                acc[0][t] = __builtin_amdgcn_wmma_f32_16x16x32_bf16(
                    false, a[0].v, false, bfr.v, (short)0, acc[0][t], false, false);
                acc[1][t] = __builtin_amdgcn_wmma_f32_16x16x32_bf16(
                    false, a[1].v, false, bfr.v, (short)0, acc[1][t], false, false);
            }
        }
        wait_async_zero();
        __syncthreads();
    }

    // C/D layout: VGPR v -> row (hi*8 + v); lane%16 -> column
#pragma unroll
    for (int mt = 0; mt < 2; ++mt) {
#pragma unroll
        for (int t = 0; t < 4; ++t) {
            const int n  = nc[t];
            const float bv = bias[n];
#pragma unroll
            for (int v = 0; v < 8; ++v) {
                int row = blockM + mwave * 32 + mt * 16 + hi * 8 + v;
                float val = acc[mt][t][v] + bv;
                if (ACT == 1) val = fmaxf(val, 0.0f);
                out[(long long)row * N + n] = f2bf(val);
            }
        }
    }
}

// ---------------------------------------------------------------------------
// Attention per (b,h): softmax(QK^T/sqrt(hd) + bias, mask) @ V   (VALU; tiny)
// One wave32 per block; thread = query index.
// ---------------------------------------------------------------------------
__global__ __launch_bounds__(32) void attention_kernel(
    const unsigned short* __restrict__ qkv,   // [M_TOTAL][3*D_MODEL] bf16
    const float* __restrict__ db,             // [B][H][K]
    const unsigned char* __restrict__ mask,   // [B][K]
    unsigned short* __restrict__ ctx)         // [M_TOTAL][D_MODEL] bf16
{
    __shared__ unsigned short Qs[NK * HEAD_DIM];
    __shared__ unsigned short Ks[NK * HEAD_DIM];
    __shared__ unsigned short Vs[NK * HEAD_DIM];

    const int bh = blockIdx.x;
    const int b  = bh >> 3;
    const int h  = bh & 7;
    const int q  = threadIdx.x;   // 0..31

    const long long rs   = 3LL * D_MODEL;
    const long long base = (long long)b * NK * rs + (long long)h * HEAD_DIM;

    {
        const uint4* qr = (const uint4*)(qkv + base + (long long)q * rs);
        const uint4* kr = (const uint4*)(qkv + base + (long long)q * rs + D_MODEL);
        const uint4* vr = (const uint4*)(qkv + base + (long long)q * rs + 2 * D_MODEL);
        uint4* qd = (uint4*)(Qs + q * HEAD_DIM);
        uint4* kd = (uint4*)(Ks + q * HEAD_DIM);
        uint4* vd = (uint4*)(Vs + q * HEAD_DIM);
#pragma unroll
        for (int c = 0; c < HEAD_DIM / 8; ++c) { qd[c] = qr[c]; kd[c] = kr[c]; vd[c] = vr[c]; }
    }
    __syncthreads();

    float scores[NK];
#pragma unroll
    for (int k = 0; k < NK; ++k) scores[k] = 0.0f;

    for (int d0 = 0; d0 < HEAD_DIM; d0 += 32) {
        float qr[32];
#pragma unroll
        for (int j = 0; j < 32; ++j) qr[j] = bf2f(Qs[q * HEAD_DIM + d0 + j]);
        for (int k = 0; k < NK; ++k) {
            float s = 0.0f;
#pragma unroll
            for (int j = 0; j < 32; ++j) s += qr[j] * bf2f(Ks[k * HEAD_DIM + d0 + j]);
            scores[k] += s;
        }
    }

    const float scale = rsqrtf((float)HEAD_DIM);
    const float biasq = db[((long long)b * NUM_HEADS + h) * NK + q];
    float mx = -3.4e38f;
#pragma unroll
    for (int k = 0; k < NK; ++k) {
        float s = scores[k] * scale + biasq;
        if (!mask[b * NK + k]) s = -10000.0f;
        scores[k] = s;
        mx = fmaxf(mx, s);
    }
    float sum = 0.0f;
#pragma unroll
    for (int k = 0; k < NK; ++k) { float e = __expf(scores[k] - mx); scores[k] = e; sum += e; }
    const float inv = 1.0f / sum;

    unsigned short* crow = ctx + (long long)(b * NK + q) * D_MODEL + (long long)h * HEAD_DIM;
    for (int d = 0; d < HEAD_DIM; ++d) {
        float a = 0.0f;
#pragma unroll
        for (int k = 0; k < NK; ++k) a += scores[k] * bf2f(Vs[k * HEAD_DIM + d]);
        crow[d] = f2bf(a * inv);
    }
}

// ---------------------------------------------------------------------------
// x = LayerNorm(x + delta) * g + b ; writes f32 residual stream + bf16 mirror
// ---------------------------------------------------------------------------
__global__ __launch_bounds__(256) void residual_ln_kernel(
    float* __restrict__ x, const unsigned short* __restrict__ delta,
    const float* __restrict__ g, const float* __restrict__ bb,
    unsigned short* __restrict__ xbf)
{
    __shared__ float red[256];
    const int row = blockIdx.x;
    const int tid = threadIdx.x;
    const long long off = (long long)row * D_MODEL;

    float vals[10];
    float s = 0.0f;
#pragma unroll
    for (int j = 0; j < 10; ++j) {
        int c = j * 256 + tid;
        vals[j] = x[off + c] + bf2f(delta[off + c]);
        s += vals[j];
    }
    red[tid] = s; __syncthreads();
    for (int st = 128; st > 0; st >>= 1) { if (tid < st) red[tid] += red[tid + st]; __syncthreads(); }
    const float mean = red[0] / (float)D_MODEL; __syncthreads();

    float v = 0.0f;
#pragma unroll
    for (int j = 0; j < 10; ++j) { float d = vals[j] - mean; v += d * d; }
    red[tid] = v; __syncthreads();
    for (int st = 128; st > 0; st >>= 1) { if (tid < st) red[tid] += red[tid + st]; __syncthreads(); }
    const float rstd = rsqrtf(red[0] / (float)D_MODEL + LN_EPS);

#pragma unroll
    for (int j = 0; j < 10; ++j) {
        int c = j * 256 + tid;
        float nv = (vals[j] - mean) * rstd * g[c] + bb[c];
        x[off + c]   = nv;
        xbf[off + c] = f2bf(nv);
    }
}

// ---------------------------------------------------------------------------
// Final LN + masked mean-pool over K -> out[b][D]
// ---------------------------------------------------------------------------
__global__ __launch_bounds__(256) void final_pool_kernel(
    const float* __restrict__ x, const unsigned char* __restrict__ mask,
    const float* __restrict__ g, const float* __restrict__ bb,
    float* __restrict__ out)
{
    __shared__ float red[256];
    const int b   = blockIdx.x;
    const int tid = threadIdx.x;

    float acc[10];
#pragma unroll
    for (int j = 0; j < 10; ++j) acc[j] = 0.0f;
    float cnt = 0.0f;

    for (int k = 0; k < NK; ++k) {
        const long long off = (long long)(b * NK + k) * D_MODEL;
        float vals[10];
        float s = 0.0f;
#pragma unroll
        for (int j = 0; j < 10; ++j) {
            int c = j * 256 + tid;
            vals[j] = x[off + c];
            s += vals[j];
        }
        red[tid] = s; __syncthreads();
        for (int st = 128; st > 0; st >>= 1) { if (tid < st) red[tid] += red[tid + st]; __syncthreads(); }
        const float mean = red[0] / (float)D_MODEL; __syncthreads();

        float v = 0.0f;
#pragma unroll
        for (int j = 0; j < 10; ++j) { float d = vals[j] - mean; v += d * d; }
        red[tid] = v; __syncthreads();
        for (int st = 128; st > 0; st >>= 1) { if (tid < st) red[tid] += red[tid + st]; __syncthreads(); }
        const float rstd = rsqrtf(red[0] / (float)D_MODEL + LN_EPS); __syncthreads();

        const float m = mask[b * NK + k] ? 1.0f : 0.0f;
        cnt += m;
#pragma unroll
        for (int j = 0; j < 10; ++j) {
            int c = j * 256 + tid;
            acc[j] += m * ((vals[j] - mean) * rstd * g[c] + bb[c]);
        }
    }
    const float inv = 1.0f / fmaxf(cnt, 1.0f);
#pragma unroll
    for (int j = 0; j < 10; ++j)
        out[(long long)b * D_MODEL + j * 256 + tid] = acc[j] * inv;
}

// ---------------------------------------------------------------------------
// Orchestration
// ---------------------------------------------------------------------------
extern "C" void kernel_launch(void* const* d_in, const int* in_sizes, int n_in,
                              void* d_out, int out_size, void* d_ws, size_t ws_size,
                              hipStream_t stream)
{
    const float*         neighbors = (const float*)d_in[0];
    const float*         distances = (const float*)d_in[1];
    const unsigned char* mask      = (const unsigned char*)d_in[2];
    const float* Wqkv  = (const float*)d_in[3];
    const float* bqkv  = (const float*)d_in[4];
    const float* Wo    = (const float*)d_in[5];
    const float* bo    = (const float*)d_in[6];
    const float* Wf1   = (const float*)d_in[7];
    const float* bf1   = (const float*)d_in[8];
    const float* Wf2   = (const float*)d_in[9];
    const float* bf2   = (const float*)d_in[10];
    const float* ln1_g = (const float*)d_in[11];
    const float* ln1_b = (const float*)d_in[12];
    const float* ln2_g = (const float*)d_in[13];
    const float* ln2_b = (const float*)d_in[14];
    const float* dbW1  = (const float*)d_in[15];
    const float* dbb1  = (const float*)d_in[16];
    const float* dbW2  = (const float*)d_in[17];
    const float* dbb2  = (const float*)d_in[18];
    const float* norm_g = (const float*)d_in[19];
    const float* norm_b = (const float*)d_in[20];

    char* ws = (char*)d_ws;
    size_t off = 0;
    auto alloc = [&](size_t bytes) -> void* {
        void* p = ws + off;
        off += (bytes + 255) & ~(size_t)255;
        return p;
    };

    float*          Xf    = (float*)         alloc((size_t)M_TOTAL * D_MODEL * 4);
    unsigned short* Xbf   = (unsigned short*)alloc((size_t)M_TOTAL * D_MODEL * 2);
    unsigned short* qkvB  = (unsigned short*)alloc((size_t)M_TOTAL * 3 * D_MODEL * 2);
    unsigned short* ctxB  = (unsigned short*)alloc((size_t)M_TOTAL * D_MODEL * 2);
    unsigned short* hidB  = (unsigned short*)alloc((size_t)M_TOTAL * D_FF * 2);
    unsigned short* delB  = (unsigned short*)alloc((size_t)M_TOTAL * D_MODEL * 2);
    float*          dbB   = (float*)         alloc((size_t)NB * NUM_HEADS * NK * 4);
    unsigned short* WqkvT[NUM_LAYERS], *WoT[NUM_LAYERS], *Wf1T[NUM_LAYERS], *Wf2T[NUM_LAYERS];
    for (int l = 0; l < NUM_LAYERS; ++l) {
        WqkvT[l] = (unsigned short*)alloc((size_t)3 * D_MODEL * D_MODEL * 2);
        WoT[l]   = (unsigned short*)alloc((size_t)D_MODEL * D_MODEL * 2);
        Wf1T[l]  = (unsigned short*)alloc((size_t)D_MODEL * D_FF * 2);
        Wf2T[l]  = (unsigned short*)alloc((size_t)D_FF * D_MODEL * 2);
    }

    // ---- X init ----
    {
        long long n = (long long)M_TOTAL * D_MODEL;
        cast_x_kernel<<<(unsigned)((n + 255) / 256), 256, 0, stream>>>(neighbors, Xf, Xbf, n);
    }

    // ---- weight casts (bf16 + transpose) ----
    for (int l = 0; l < NUM_LAYERS; ++l) {
        long long n1 = (long long)D_MODEL * 3 * D_MODEL;
        wcast_kernel<<<(unsigned)((n1 + 255) / 256), 256, 0, stream>>>(
            Wqkv + (long long)l * n1, WqkvT[l], D_MODEL, 3 * D_MODEL);
        long long n2 = (long long)D_MODEL * D_MODEL;
        wcast_kernel<<<(unsigned)((n2 + 255) / 256), 256, 0, stream>>>(
            Wo + (long long)l * n2, WoT[l], D_MODEL, D_MODEL);
        long long n3 = (long long)D_MODEL * D_FF;
        wcast_kernel<<<(unsigned)((n3 + 255) / 256), 256, 0, stream>>>(
            Wf1 + (long long)l * n3, Wf1T[l], D_MODEL, D_FF);
        wcast_kernel<<<(unsigned)((n3 + 255) / 256), 256, 0, stream>>>(
            Wf2 + (long long)l * n3, Wf2T[l], D_FF, D_MODEL);
    }

    // ---- distance-bias MLP ----
    dist_bias_kernel<<<(NB * NK) / 256, 256, 0, stream>>>(
        distances, dbW1, dbb1, dbW2, dbb2, dbB);

    // ---- encoder layers ----
    for (int l = 0; l < NUM_LAYERS; ++l) {
        // qkv = X @ Wqkv + bqkv
        gemm_bf16_wmma<0><<<dim3(3 * D_MODEL / GBN, M_TOTAL / GBM), 256, 0, stream>>>(
            Xbf, WqkvT[l], bqkv + (long long)l * 3 * D_MODEL, qkvB,
            M_TOTAL, 3 * D_MODEL, D_MODEL);

        // attention per (b,h)
        attention_kernel<<<NB * NUM_HEADS, 32, 0, stream>>>(qkvB, dbB, mask, ctxB);

        // attn_out = ctx @ Wo + bo
        gemm_bf16_wmma<0><<<dim3(D_MODEL / GBN, M_TOTAL / GBM), 256, 0, stream>>>(
            ctxB, WoT[l], bo + (long long)l * D_MODEL, delB,
            M_TOTAL, D_MODEL, D_MODEL);

        // x = LN(x + attn_out)
        residual_ln_kernel<<<M_TOTAL, 256, 0, stream>>>(
            Xf, delB, ln1_g + (long long)l * D_MODEL, ln1_b + (long long)l * D_MODEL, Xbf);

        // h = relu(x @ Wf1 + bf1)
        gemm_bf16_wmma<1><<<dim3(D_FF / GBN, M_TOTAL / GBM), 256, 0, stream>>>(
            Xbf, Wf1T[l], bf1 + (long long)l * D_FF, hidB,
            M_TOTAL, D_FF, D_MODEL);

        // ffn = h @ Wf2 + bf2
        gemm_bf16_wmma<0><<<dim3(D_MODEL / GBN, M_TOTAL / GBM), 256, 0, stream>>>(
            hidB, Wf2T[l], bf2 + (long long)l * D_MODEL, delB,
            M_TOTAL, D_MODEL, D_FF);

        // x = LN(x + ffn)
        residual_ln_kernel<<<M_TOTAL, 256, 0, stream>>>(
            Xf, delB, ln2_g + (long long)l * D_MODEL, ln2_b + (long long)l * D_MODEL, Xbf);
    }

    // ---- final LN + masked mean-pool ----
    final_pool_kernel<<<NB, 256, 0, stream>>>(Xf, mask, norm_g, norm_b, (float*)d_out);
}
